// MultiHeadAttention_18837726560936
// MI455X (gfx1250) — matrix operand order repaired and checked by reading, not compiled, not run on hardware
//
#include <hip/hip_runtime.h>
#include <hip/hip_bf16.h>
#include <math.h>

#define DEVFN __device__ __forceinline__

typedef __attribute__((ext_vector_type(16))) __bf16 v16bf;
typedef __attribute__((ext_vector_type(8)))  float  v8f;

constexpr int   cL   = 2048;
constexpr int   cB   = 4;
constexpr int   cHID = 512;
constexpr int   cNH  = 8;
constexpr int   cD   = 64;     // head dim
constexpr int   cLQ  = 128;
constexpr int   cBK  = cB * cNH;             // 32 heads total
constexpr float cINV_SCALE = 0.125f;         // 1/sqrt(64)

union Frag2 { uint4 u[2]; v16bf v; };
union U16x8 { uint4 u; __bf16 b[8]; };

DEVFN v8f zero8() {
  v8f z;
#pragma unroll
  for (int e = 0; e < 8; ++e) z[e] = 0.0f;
  return z;
}

DEVFN v8f wmma_bf16(v16bf a, v16bf b, v8f c) {
  // D = A(16x32 bf16) x B(32x16 bf16) + C(16x16 f32)
  return __builtin_amdgcn_wmma_f32_16x16x32_bf16(false, a, false, b, (short)0, c,
                                                 false, false);
}

// CDNA5 async global->LDS copy (ASYNCcnt-tracked), 16B per lane.
DEVFN void async_copy_b128(const __bf16* lds_dst, const __bf16* gsrc) {
  unsigned ldsoff = (unsigned)(uintptr_t)lds_dst;    // low 32 bits = LDS offset
  unsigned long long ga = (unsigned long long)(uintptr_t)gsrc;
  asm volatile("global_load_async_to_lds_b128 %0, %1, off"
               :: "v"(ldsoff), "v"(ga)
               : "memory");
}
DEVFN void wait_async0() {
  asm volatile("s_wait_asynccnt 0x0" ::: "memory");
}

// ---- fragment builders (contiguous k order, same in A and B) -----------------------
// A 16x32 bf16: lane = row (lane&15); element e -> K = h*16 + e
DEVFN v16bf afrag_bf16(const __bf16* p, int h) {
  Frag2 f;
  f.u[0] = *(const uint4*)(p + h * 16);       // K = h*16 .. h*16+7
  f.u[1] = *(const uint4*)(p + h * 16 + 8);   // K = h*16+8 .. h*16+15
  return f.v;
}
DEVFN v16bf afrag_f32(const float* p, int h) {
  v16bf v;
#pragma unroll
  for (int e = 0; e < 8; ++e) {
    v[e]     = (__bf16)p[h * 16 + e];
    v[8 + e] = (__bf16)p[h * 16 + 8 + e];
  }
  return v;
}
// B 32x16 bf16: lane = col (lane&15); element e -> K = h*16 + e
DEVFN v16bf bfrag_bf16(const __bf16* p, int h) {
  Frag2 f;
  f.u[0] = *(const uint4*)(p + h * 16);
  f.u[1] = *(const uint4*)(p + h * 16 + 8);
  return f.v;
}
DEVFN v16bf bfrag_f32(const float* p, int h) {
  v16bf v;
#pragma unroll
  for (int e = 0; e < 16; ++e) v[e] = (__bf16)p[h * 16 + e];
  return v;
}
DEVFN v16bf bfrag_f32s(const float* p, int stride, int h) {
  v16bf v;
#pragma unroll
  for (int e = 0; e < 16; ++e) v[e] = (__bf16)p[(unsigned)(h * 16 + e) * stride];
  return v;
}

// 16-lane (half-wave) reductions: rows of a C-tile live in one 16-lane half.
DEVFN float halfmax(float v) {
  v = fmaxf(v, __shfl_xor(v, 1, 32));
  v = fmaxf(v, __shfl_xor(v, 2, 32));
  v = fmaxf(v, __shfl_xor(v, 4, 32));
  v = fmaxf(v, __shfl_xor(v, 8, 32));
  return v;
}
DEVFN float halfsum(float v) {
  v += __shfl_xor(v, 1, 32);
  v += __shfl_xor(v, 2, 32);
  v += __shfl_xor(v, 4, 32);
  v += __shfl_xor(v, 8, 32);
  return v;
}

// ---- kernel 0: log1p(-mask), clamped ---------------------------------------
__global__ void k_lmask(const float* __restrict__ kpm, float* __restrict__ lm) {
  int i = blockIdx.x * blockDim.x + threadIdx.x;
  if (i < cB * cL) {
    float v = log1pf(-kpm[i]);
    if (!(v > -1.0e30f)) v = -1.0e30f;
    lm[i] = v;
  }
}

// ---- kernel 1: projection GEMM  Y = X @ W^T -> bf16 head layout ------------
__global__ void __launch_bounds__(256) k_proj(const float* __restrict__ X,
                                              const float* __restrict__ W,
                                              __bf16* __restrict__ outh) {
  const int lane = threadIdx.x & 31, ln = lane & 15, h = lane >> 4;
  const int wave = threadIdx.x >> 5;
  const int tile = blockIdx.x * 8 + wave;       // 16384 tiles
  const int m0 = (tile >> 5) * 16;
  const int n0 = (tile & 31) * 16;
  const float* arow = X + (unsigned)(m0 + ln) * cHID;
  const float* brow = W + (unsigned)(n0 + ln) * cHID;
  v8f acc = zero8();
  for (int kk = 0; kk < cHID; kk += 32)
    acc = wmma_bf16(afrag_f32(arow + kk, h), bfrag_f32(brow + kk, h), acc);
#pragma unroll
  for (int r = 0; r < 8; ++r) {
    int m = m0 + r + 8 * h;                     // m = l*B + b
    int n = n0 + ln;
    int l = m >> 2, b = m & 3;
    int head = b * cNH + (n >> 6);
    outh[(unsigned)(head * cL + l) * cD + (n & 63)] = (__bf16)acc[r];
  }
}

// ---- kernel 2: t[n,i,h'] = sum_d q[n, L-LQ+i, d] * Wk[(n&7)*64+d, h'] -------
__global__ void __launch_bounds__(256) k_t(const __bf16* __restrict__ qh,
                                           const float* __restrict__ Wk,
                                           __bf16* __restrict__ t_ws) {
  const int lane = threadIdx.x & 31, ln = lane & 15, h = lane >> 4;
  const int wave = threadIdx.x >> 5;
  const int tile = blockIdx.x * 8 + wave;       // 8192 tiles
  const int n = tile >> 8;                      // 8 i-tiles * 32 h-tiles per head
  const int rem = tile & 255;
  const int i0 = (rem >> 5) * 16;
  const int h0 = (rem & 31) * 16;
  const int khead = n & 7;
  const __bf16* ap = qh + (unsigned)(n * cL + (cL - cLQ) + i0 + ln) * cD;
  v8f acc = zero8();
#pragma unroll
  for (int kk = 0; kk < cD; kk += 32)
    acc = wmma_bf16(afrag_bf16(ap + kk, h),
                    bfrag_f32s(Wk + (unsigned)(khead * cD + kk) * cHID + h0 + ln,
                               cHID, h),
                    acc);
#pragma unroll
  for (int r = 0; r < 8; ++r)
    t_ws[(unsigned)(n * cLQ + i0 + r + 8 * h) * cHID + h0 + ln] = (__bf16)acc[r];
}

// ---- kernel 3: bias[n,i,j] = sum_h qre[i,j,b,h] * t[n,i,h] ------------------
__global__ void __launch_bounds__(256) k_bias(const float* __restrict__ qre,
                                              const __bf16* __restrict__ t_ws,
                                              float* __restrict__ bias) {
  const int lane = threadIdx.x & 31, ln = lane & 15, h = lane >> 4;
  const int wave = threadIdx.x >> 5;            // j-tile
  const int b = blockIdx.x >> 7;
  const int i = blockIdx.x & 127;
  const int j0 = wave * 16;
  const float*  ap = qre + ((unsigned)(i * cLQ + j0 + ln) * cB + b) * cHID;
  const __bf16* bp = t_ws + (unsigned)((b * cNH + (ln & 7)) * cLQ + i) * cHID;
  v8f acc = zero8();
  for (int kk = 0; kk < cHID; kk += 32)
    acc = wmma_bf16(afrag_f32(ap + kk, h), bfrag_bf16(bp + kk, h), acc);
  if (ln < 8) {
#pragma unroll
    for (int r = 0; r < 8; ++r)
      bias[(unsigned)((b * cNH + ln) * cLQ + i) * cLQ + j0 + r + 8 * h] = acc[r];
  }
}

// ---- kernel 4: two-pass flash attention -------------------------------------
// Pass 1: online row stats (m,l) only.  Pass 2: recompute S, emit normalized
// probabilities straight to attn (touched exactly once) and accumulate O = P.V.
__global__ void __launch_bounds__(128) k_attn(
    const __bf16* __restrict__ qh, const __bf16* __restrict__ kh,
    const __bf16* __restrict__ vh, const float* __restrict__ lmask,
    const float* __restrict__ bias, float* __restrict__ attn,
    __bf16* __restrict__ oh) {
  __shared__ alignas(16) __bf16 kbuf[32 * cD];      // [j][d], row major
  __shared__ alignas(16) __bf16 vbufT[cD * 32];     // [d][j], transposed
  __shared__ alignas(16) __bf16 ptile[4][16 * 32];  // per-wave P transpose
  const int tid = threadIdx.x;
  const int lane = tid & 31, ln = lane & 15, h = lane >> 4;
  const int wave = tid >> 5;
  const int n  = blockIdx.x >> 5;                   // head
  const int rb = blockIdx.x & 31;                   // row block
  const int bb = n >> 3;                            // batch
  const int q0 = rb * 64 + wave * 16;
  const __bf16* qrow = qh + (unsigned)(n * cL + q0 + ln) * cD;
  v16bf qa0 = afrag_bf16(qrow, h);
  v16bf qa1 = afrag_bf16(qrow + 32, h);
  float mst[8], lst[8];
#pragma unroll
  for (int r = 0; r < 8; ++r) { mst[r] = -3.0e38f; lst[r] = 0.0f; }

  // ---------------- pass 1: stats ----------------
  for (int j0 = 0; j0 < cL; j0 += 32) {
    __syncthreads();
    const __bf16* ksrc = kh + (unsigned)(n * cL + j0) * cD;
    async_copy_b128(kbuf + tid * 8, ksrc + tid * 8);
    async_copy_b128(kbuf + (tid + 128) * 8, ksrc + (tid + 128) * 8);
    if (j0 + 32 < cL) __builtin_prefetch(ksrc + 32 * cD, 0, 0);
    wait_async0();
    __syncthreads();

    v8f s0a = zero8(), s1a = zero8();
    s0a = wmma_bf16(qa0, bfrag_bf16(kbuf + ln * cD, h), s0a);
    s0a = wmma_bf16(qa1, bfrag_bf16(kbuf + ln * cD + 32, h), s0a);
    s1a = wmma_bf16(qa0, bfrag_bf16(kbuf + (16 + ln) * cD, h), s1a);
    s1a = wmma_bf16(qa1, bfrag_bf16(kbuf + (16 + ln) * cD + 32, h), s1a);

    float lm0 = lmask[bb * cL + j0 + ln];
    float lm1 = lmask[bb * cL + j0 + 16 + ln];
#pragma unroll
    for (int r = 0; r < 8; ++r) {
      int q = q0 + r + 8 * h;
      float d0 = s0a[r], d1 = s1a[r];
      int qq = q - (cL - cLQ);
      if (qq >= 0) {
        int jj0 = j0 + ln - (cL - cLQ);
        if (jj0 >= 0)      d0 += bias[(unsigned)(n * cLQ + qq) * cLQ + jj0];
        if (jj0 + 16 >= 0) d1 += bias[(unsigned)(n * cLQ + qq) * cLQ + jj0 + 16];
      }
      float s0 = d0 * cINV_SCALE + lm0;
      float s1 = d1 * cINV_SCALE + lm1;
      float mnew = fmaxf(mst[r], halfmax(fmaxf(s0, s1)));
      float sc = __expf(mst[r] - mnew);
      float pp = __expf(s0 - mnew) + __expf(s1 - mnew);
      lst[r] = lst[r] * sc + halfsum(pp);
      mst[r] = mnew;
    }
  }
  float linv[8];
#pragma unroll
  for (int r = 0; r < 8; ++r) linv[r] = 1.0f / lst[r];

  // ---------------- pass 2: emit P, accumulate O ----------------
  v8f o0 = zero8(), o1 = zero8(), o2 = zero8(), o3 = zero8();
  __bf16* ptw = ptile[wave];
  for (int j0 = 0; j0 < cL; j0 += 32) {
    __syncthreads();
    const __bf16* ksrc = kh + (unsigned)(n * cL + j0) * cD;
    const __bf16* vsrc = vh + (unsigned)(n * cL + j0) * cD;
    async_copy_b128(kbuf + tid * 8, ksrc + tid * 8);
    async_copy_b128(kbuf + (tid + 128) * 8, ksrc + (tid + 128) * 8);
#pragma unroll
    for (int part = 0; part < 2; ++part) {
      int idx = tid + part * 128;
      U16x8 raw; raw.u = ((const uint4*)vsrc)[idx];
      int j = idx >> 3;                 // 8 uint4 per 64-elem row
      int dbase = (idx & 7) * 8;
#pragma unroll
      for (int e = 0; e < 8; ++e) vbufT[(dbase + e) * 32 + j] = raw.b[e];
    }
    if (j0 + 32 < cL) __builtin_prefetch(vsrc + 32 * cD, 0, 0);
    wait_async0();
    __syncthreads();

    v8f s0a = zero8(), s1a = zero8();
    s0a = wmma_bf16(qa0, bfrag_bf16(kbuf + ln * cD, h), s0a);
    s0a = wmma_bf16(qa1, bfrag_bf16(kbuf + ln * cD + 32, h), s0a);
    s1a = wmma_bf16(qa0, bfrag_bf16(kbuf + (16 + ln) * cD, h), s1a);
    s1a = wmma_bf16(qa1, bfrag_bf16(kbuf + (16 + ln) * cD + 32, h), s1a);

    float lm0 = lmask[bb * cL + j0 + ln];
    float lm1 = lmask[bb * cL + j0 + 16 + ln];
    float p0[8], p1[8];
#pragma unroll
    for (int r = 0; r < 8; ++r) {
      int q = q0 + r + 8 * h;
      float d0 = s0a[r], d1 = s1a[r];
      int qq = q - (cL - cLQ);
      if (qq >= 0) {
        int jj0 = j0 + ln - (cL - cLQ);
        if (jj0 >= 0)      d0 += bias[(unsigned)(n * cLQ + qq) * cLQ + jj0];
        if (jj0 + 16 >= 0) d1 += bias[(unsigned)(n * cLQ + qq) * cLQ + jj0 + 16];
      }
      float s0 = d0 * cINV_SCALE + lm0;
      float s1 = d1 * cINV_SCALE + lm1;
      p0[r] = __expf(s0 - mst[r]) * linv[r];     // final normalized prob
      p1[r] = __expf(s1 - mst[r]) * linv[r];
      unsigned abase = (unsigned)(n * cL + q) * (unsigned)cL + j0 + ln;
      attn[abase]      = p0[r];
      attn[abase + 16] = p1[r];
    }
    // transpose P through LDS into A-frag layout
#pragma unroll
    for (int r = 0; r < 8; ++r) {
      ptw[(r + 8 * h) * 32 + ln]      = (__bf16)p0[r];
      ptw[(r + 8 * h) * 32 + 16 + ln] = (__bf16)p1[r];
    }
    v16bf pa = afrag_bf16(ptw + ln * 32, h);
    o0 = wmma_bf16(pa, bfrag_bf16(vbufT + ln * 32, h), o0);
    o1 = wmma_bf16(pa, bfrag_bf16(vbufT + (16 + ln) * 32, h), o1);
    o2 = wmma_bf16(pa, bfrag_bf16(vbufT + (32 + ln) * 32, h), o2);
    o3 = wmma_bf16(pa, bfrag_bf16(vbufT + (48 + ln) * 32, h), o3);
  }

#pragma unroll
  for (int r = 0; r < 8; ++r) {
    int q = q0 + r + 8 * h;
    __bf16* orow = oh + (unsigned)(n * cL + q) * cD;
    orow[ln]      = (__bf16)o0[r];
    orow[16 + ln] = (__bf16)o1[r];
    orow[32 + ln] = (__bf16)o2[r];
    orow[48 + ln] = (__bf16)o3[r];
  }
}

// ---- kernel 5: out[l,b,:] = (concat heads) @ Wo^T ---------------------------
__global__ void __launch_bounds__(256) k_out(const __bf16* __restrict__ oh,
                                             const float* __restrict__ Wo,
                                             float* __restrict__ out) {
  const int lane = threadIdx.x & 31, ln = lane & 15, h = lane >> 4;
  const int wave = threadIdx.x >> 5;
  const int tile = blockIdx.x * 8 + wave;
  const int m0 = (tile >> 5) * 16;
  const int n0 = (tile & 31) * 16;
  const int row = m0 + ln;
  const int l = row >> 2, b = row & 3;
  v8f acc = zero8();
  for (int kk = 0; kk < cHID; kk += 32) {
    const __bf16* ap =
        oh + (unsigned)((b * cNH + (kk >> 6)) * cL + l) * cD + (kk & 63);
    acc = wmma_bf16(afrag_bf16(ap, h),
                    bfrag_f32(Wo + (unsigned)(n0 + ln) * cHID + kk, h), acc);
  }
#pragma unroll
  for (int r = 0; r < 8; ++r)
    out[(unsigned)(m0 + r + 8 * h) * cHID + n0 + ln] = acc[r];
}

// ---- launcher ---------------------------------------------------------------
extern "C" void kernel_launch(void* const* d_in, const int* in_sizes, int n_in,
                              void* d_out, int out_size, void* d_ws,
                              size_t ws_size, hipStream_t stream) {
  const float* query = (const float*)d_in[0];
  const float* key_t = (const float*)d_in[1];
  const float* value = (const float*)d_in[2];
  const float* kpm   = (const float*)d_in[3];
  const float* qre   = (const float*)d_in[4];
  const float* Wq    = (const float*)d_in[5];
  const float* Wk    = (const float*)d_in[6];
  const float* Wv    = (const float*)d_in[7];
  const float* Wo    = (const float*)d_in[8];

  float* out  = (float*)d_out;                                // [L,B,HID]
  float* attn = out + (size_t)cL * cB * cHID;                 // [BK,L,L]

  constexpr size_t headBytes = (size_t)cBK * cL * cD * 2;     // 8 MB (bf16)
  char* ws = (char*)d_ws;
  __bf16* qh   = (__bf16*)(ws);
  __bf16* kh   = (__bf16*)(ws + headBytes);
  __bf16* vh   = (__bf16*)(ws + 2 * headBytes);
  __bf16* ohb  = (__bf16*)(ws + 3 * headBytes);
  __bf16* t_ws = (__bf16*)(ws + 4 * headBytes);               // 4 MB
  float*  bias = (float*) (ws + 4 * headBytes + (size_t)cBK * cLQ * cHID * 2);
  float*  lmsk = bias + (size_t)cBK * cLQ * cLQ;

  k_lmask<<<(cB * cL + 255) / 256, 256, 0, stream>>>(kpm, lmsk);

  k_proj<<<2048, 256, 0, stream>>>(query, Wq, qh);
  k_proj<<<2048, 256, 0, stream>>>(key_t, Wk, kh);
  k_proj<<<2048, 256, 0, stream>>>(value, Wv, vh);

  k_t<<<1024, 256, 0, stream>>>(qh, Wk, t_ws);
  k_bias<<<512, 256, 0, stream>>>(qre, t_ws, bias);

  k_attn<<<cBK * 32, 128, 0, stream>>>(qh, kh, vh, lmsk, bias, attn, ohb);

  k_out<<<2048, 256, 0, stream>>>(ohb, Wo, out);
}
